// GNN_3865470567039
// MI455X (gfx1250) — matrix operand order repaired and checked by reading, not compiled, not run on hardware
//
#include <hip/hip_runtime.h>
#include <hip/hip_bf16.h>

// ---------------------------------------------------------------------------
// AttentiveFP-style GNN pipeline on gfx1250 (wave32, WMMA).
// All dense GEMMs -> v_wmma_f32_16x16x32_f16, f32 accumulate.
// Weights pre-packed to B-fragment layout so the inner loop is
// 2x ds_load_b128 (A) + 2x global_load_b128 (B) + v_wmma, unrolled over K.
// ---------------------------------------------------------------------------

typedef __attribute__((ext_vector_type(16))) _Float16 v16h;
typedef __attribute__((ext_vector_type(8)))  _Float16 v8h;
typedef __attribute__((ext_vector_type(8)))  float    v8f;

#define LREL(v) ((v) > 0.f ? (v) : 0.01f * (v))

__device__ __forceinline__ unsigned ford(float f) {
  unsigned u = __float_as_uint(f);
  return (u & 0x80000000u) ? ~u : (u | 0x80000000u);
}
__device__ __forceinline__ float funord(unsigned u) {
  return (u & 0x80000000u) ? __uint_as_float(u & 0x7fffffffu) : __uint_as_float(~u);
}

// ---------------------------------------------------------------------------
// Pack f32 weights [K x COUT] into B-fragment layout, zero-padded to KP:
//   Wp[((cb*NK + t)*32 + lane)*16 + i] = W[t*32 + 16*(lane>>4) + i, cb*16 + (lane&15)]
// ---------------------------------------------------------------------------
__global__ void k_packw(const float* __restrict__ W, _Float16* __restrict__ Wp,
                        int K, int KP, int COUT)
{
  int nk = KP / 32;
  long total = (long)(COUT / 16) * nk * 512;
  long idx = blockIdx.x * (long)blockDim.x + threadIdx.x;
  if (idx >= total) return;
  int i  = (int)(idx & 15);
  int l  = (int)((idx >> 4) & 31);
  int t  = (int)((idx >> 9) % nk);
  int cb = (int)(idx / (512L * nk));
  int k   = t * 32 + ((l >> 4) << 4) + i;
  int col = (cb << 4) + (l & 15);
  Wp[idx] = (k < K) ? (_Float16)W[(long)k * COUT + col] : (_Float16)0.f;
}

// ---------------------------------------------------------------------------
// WMMA GEMM: Y[M x COUT] = act(X[M x K] @ W + bias)
// XMODE: 0 = f32 X, 1 = f16 X, 2 = concat( x[gatherIdx[row]] (128 f32), eattr (16 f32) )
// Output: Yh f16 store | scatterIdx: atomicAdd scale[row]*val into Yf rows | Yf f32.
// Block = 128 threads = 4 waves; wave computes 16 rows x 32 cols; grid.y = COUT/128.
// REQUIRES: M % 16 == 0 (true for all call sites).
// ---------------------------------------------------------------------------
template<int K, int XMODE>
__global__ __launch_bounds__(128)
void k_wmma_gemm(const void* __restrict__ Xsrc,
                 const int* __restrict__ gatherIdx, const float* __restrict__ eattr,
                 const _Float16* __restrict__ Wp, const float* __restrict__ bias,
                 float* __restrict__ Yf, _Float16* __restrict__ Yh,
                 const int* __restrict__ scatterIdx, const float* __restrict__ scale,
                 int M, int COUT, int act)
{
  constexpr int KP = (K + 31) & ~31;
  constexpr int NK = KP / 32;
  __shared__ _Float16 sA[16 * KP];

  const int rowbase = blockIdx.x * 16;
  const int wave    = threadIdx.x >> 5;
  const int lane    = threadIdx.x & 31;
  const int colbase = blockIdx.y * 128 + wave * 32;
  const int l15     = lane & 15;
  const int g       = lane >> 4;

  // ---- stage A tile (16 x KP halfs) into LDS, vectorized ----
  if (XMODE == 0) {
    constexpr int C4 = K / 4;
    const float4* Xv = (const float4*)Xsrc;
    for (int idx = threadIdx.x; idx < 16 * C4; idx += 128) {
      int r = idx / C4, s = idx - r * C4;
      float4 v = Xv[(long)(rowbase + r) * C4 + s];
      _Float16* d = &sA[r * KP + s * 4];
      d[0] = (_Float16)v.x; d[1] = (_Float16)v.y;
      d[2] = (_Float16)v.z; d[3] = (_Float16)v.w;
    }
  } else if (XMODE == 1) {
    constexpr int C8 = K / 8;
    const v8h* Xv = (const v8h*)Xsrc;
    for (int idx = threadIdx.x; idx < 16 * C8; idx += 128) {
      int r = idx / C8, s = idx - r * C8;
      *(v8h*)&sA[r * KP + s * 8] = Xv[(long)(rowbase + r) * C8 + s];
    }
  } else {                       // concat(x[src] 128 f32, eattr 16 f32), KP = 160
    constexpr int S4 = KP / 4;   // 40 float4 slots per row
    for (int idx = threadIdx.x; idx < 16 * S4; idx += 128) {
      int r = idx / S4, s = idx - r * S4;
      int row = rowbase + r;
      float4 v;
      if (s < 32)      v = ((const float4*)Xsrc)[(long)gatherIdx[row] * 32 + s];
      else if (s < 36) v = ((const float4*)eattr)[(long)row * 4 + (s - 32)];
      else             v = make_float4(0.f, 0.f, 0.f, 0.f);
      _Float16* d = &sA[r * KP + s * 4];
      d[0] = (_Float16)v.x; d[1] = (_Float16)v.y;
      d[2] = (_Float16)v.z; d[3] = (_Float16)v.w;
    }
  }
  __syncthreads();

  // ---- K-unrolled WMMA loop: one A fragment feeds two 16x16 col tiles ----
  const int rowa = l15 * KP;
  const int cb0  = colbase >> 4;
  v8f acc0 = {}, acc1 = {};
#pragma unroll
  for (int t = 0; t < NK; ++t) {
    const int k0 = t * 32;
    v16h a;
#pragma unroll
    for (int i = 0; i < 16; ++i)           // one contiguous 16-half chunk
      a[i] = sA[rowa + k0 + 16 * g + i];
    const v16h b0 = *(const v16h*)&Wp[(((long)cb0       * NK + t) * 32 + lane) * 16];
    const v16h b1 = *(const v16h*)&Wp[(((long)(cb0 + 1) * NK + t) * 32 + lane) * 16];
    acc0 = __builtin_amdgcn_wmma_f32_16x16x32_f16(false, a, false, b0,
                                                  (short)0, acc0, false, false);
    acc1 = __builtin_amdgcn_wmma_f32_16x16x32_f16(false, a, false, b1,
                                                  (short)0, acc1, false, false);
  }

  // ---- epilogue (M % 16 == 0 -> no row guard) ----
#pragma unroll
  for (int h = 0; h < 2; ++h) {
    const int col = colbase + h * 16 + l15;
#pragma unroll
    for (int j = 0; j < 8; ++j) {
      int row = rowbase + j + 8 * g;
      float v = h ? acc1[j] : acc0[j];
      if (bias) v += bias[col];
      if (act == 1)      v = LREL(v);
      else if (act == 2) v = fmaxf(v, 0.f);
      if (Yh)              Yh[(long)row * COUT + col] = (_Float16)v;
      else if (scatterIdx) atomicAdd(&Yf[(long)scatterIdx[row] * COUT + col],
                                     scale[row] * v);
      else                 Yf[(long)row * COUT + col] = v;
    }
  }
}

// ---- row dot: out[m] = X[m,:] . v  (+ bias[0]) ; one wave per row ----
__global__ void k_rowdot(const void* __restrict__ X, int isHalf,
                         const float* __restrict__ v, const float* __restrict__ bias,
                         float* __restrict__ out, int M, int K)
{
  int w = (int)((blockIdx.x * (long)blockDim.x + threadIdx.x) >> 5);
  int lane = threadIdx.x & 31;
  if (w >= M) return;
  float s = 0.f;
  if (isHalf) {
    const _Float16* Xh = (const _Float16*)X;
    for (int k = lane; k < K; k += 32) s += (float)Xh[(long)w * K + k] * v[k];
  } else {
    const float* Xf = (const float*)X;
    for (int k = lane; k < K; k += 32) s += Xf[(long)w * K + k] * v[k];
  }
#pragma unroll
  for (int off = 16; off > 0; off >>= 1) s += __shfl_down(s, off, 32);
  if (lane == 0) out[w] = s + (bias ? bias[0] : 0.f);
}

// ---- logit[i] = lrelu( (aidx? a[aidx[i]] : a[i]) + b[bidx[i]] ) ----
__global__ void k_logit(const float* __restrict__ a, const int* __restrict__ aidx,
                        const float* __restrict__ b, const int* __restrict__ bidx,
                        float* __restrict__ out, int n)
{
  int i = blockIdx.x * blockDim.x + threadIdx.x;
  if (i >= n) return;
  float v = (aidx ? a[aidx[i]] : a[i]) + b[bidx[i]];
  out[i] = LREL(v);
}

// ---- segment softmax pieces ----
__global__ void k_fill_f32(float* a, float v, int n) {
  int i = blockIdx.x * blockDim.x + threadIdx.x; if (i < n) a[i] = v;
}
__global__ void k_fill_u32(unsigned* a, unsigned v, int n) {
  int i = blockIdx.x * blockDim.x + threadIdx.x; if (i < n) a[i] = v;
}
__global__ void k_seg_max(const float* __restrict__ logit, const int* __restrict__ seg,
                          unsigned* __restrict__ mx, int n) {
  int i = blockIdx.x * blockDim.x + threadIdx.x;
  if (i < n) atomicMax(&mx[seg[i]], ford(logit[i]));
}
__global__ void k_seg_exp(float* __restrict__ logit, const int* __restrict__ seg,
                          const unsigned* __restrict__ mx, float* __restrict__ sum, int n) {
  int i = blockIdx.x * blockDim.x + threadIdx.x;
  if (i >= n) return;
  float v = __expf(logit[i] - funord(mx[seg[i]]));
  logit[i] = v;
  atomicAdd(&sum[seg[i]], v);
}
__global__ void k_seg_norm(float* __restrict__ logit, const int* __restrict__ seg,
                           const float* __restrict__ sum, int n) {
  int i = blockIdx.x * blockDim.x + threadIdx.x;
  if (i < n) logit[i] = logit[i] / (sum[seg[i]] + 1e-16f);
}

// ---- out[seg[r]*C + c] += (a? a[r]:1) * X[(gidx? gidx[r]:r)*C + c] ----
__global__ void k_scatter_rows(const float* __restrict__ X, const int* __restrict__ gidx,
                               const float* __restrict__ a, const int* __restrict__ seg,
                               float* __restrict__ out, int Nrows, int Cc)
{
  long t = blockIdx.x * (long)blockDim.x + threadIdx.x;
  if (t >= (long)Nrows * Cc) return;
  int r = (int)(t / Cc), c = (int)(t - (long)r * Cc);
  float v = X[(long)(gidx ? gidx[r] : r) * Cc + c];
  if (a) v *= a[r];
  atomicAdd(&out[(long)seg[r] * Cc + c], v);
}

// ---- x = relu( atl + elu(m + gb[c]) + res0? ) ----
__global__ void k_combine(const float* __restrict__ atl, const float* __restrict__ m,
                          const float* __restrict__ gb, const float* __restrict__ res0,
                          float* __restrict__ xout, long n, int Cc)
{
  long t = blockIdx.x * (long)blockDim.x + threadIdx.x;
  if (t >= n) return;
  int c = (int)(t % Cc);
  float mv = m[t] + gb[c];
  float e = mv > 0.f ? mv : (__expf(mv) - 1.f);
  float v = atl[t] + e + (res0 ? res0[t] : 0.f);
  xout[t] = fmaxf(v, 0.f);
}

__global__ void k_relu_ip(float* a, long n) {
  long t = blockIdx.x * (long)blockDim.x + threadIdx.x;
  if (t < n) a[t] = fmaxf(a[t], 0.f);
}
__global__ void k_elu_bias_ip(float* a, const float* __restrict__ b, long n, int Cc) {
  long t = blockIdx.x * (long)blockDim.x + threadIdx.x;
  if (t >= n) return;
  float v = a[t] + b[(int)(t % Cc)];
  a[t] = v > 0.f ? v : (__expf(v) - 1.f);
}
__global__ void k_gru(const float* __restrict__ gi, const float* __restrict__ gh,
                      float* __restrict__ out, int Gn, int Cc)
{
  long t = blockIdx.x * (long)blockDim.x + threadIdx.x;
  if (t >= (long)Gn * Cc) return;
  int gidx = (int)(t / Cc), c = (int)(t - (long)gidx * Cc);
  long base = (long)gidx * 3 * Cc;
  float ir = gi[base + c],        hr = gh[base + c];
  float iz = gi[base + Cc + c],   hz = gh[base + Cc + c];
  float in = gi[base + 2*Cc + c], hn = gh[base + 2*Cc + c];
  float r = 1.f / (1.f + __expf(-(ir + hr)));
  float z = 1.f / (1.f + __expf(-(iz + hz)));
  float ng = tanhf(in + r * hn);
  float o = (1.f - z) * ng + z * out[t];
  out[t] = fmaxf(o, 0.f);
}
__global__ void k_copy(const float* __restrict__ a, float* __restrict__ b, long n) {
  long t = blockIdx.x * (long)blockDim.x + threadIdx.x;
  if (t < n) b[t] = a[t];
}

// ---------------------------------------------------------------------------
extern "C" void kernel_launch(void* const* d_in, const int* in_sizes, int n_in,
                              void* d_out, int out_size, void* d_ws, size_t ws_size,
                              hipStream_t stream)
{
  constexpr int  Nn = 200000, Ee = 800000, INd = 64, Cc = 128, EDd = 16, Gg = 8000;
  constexpr long NC = (long)Nn * Cc, EC = (long)Ee * Cc, GC = (long)Gg * Cc;
  constexpr unsigned NEG_INF_ORD = 0x007FFFFFu;   // ford(-inf)

  // inputs (setup_inputs dict order)
  const float* x0        = (const float*)d_in[0];
  const int*   eidx      = (const int*)d_in[1];
  const int*   src       = eidx;
  const int*   dst       = eidx + Ee;
  const float* edge_attr = (const float*)d_in[2];
  const int*   batch     = (const int*)d_in[3];
  const float* lin1_w    = (const float*)d_in[4];
  const float* lin1_b    = (const float*)d_in[5];
  const float* gate_att_l= (const float*)d_in[6];
  const float* gate_att_r= (const float*)d_in[7];
  const float* gate_l1_w = (const float*)d_in[8];
  const float* gate_l2_w = (const float*)d_in[9];
  const float* gate_bias = (const float*)d_in[10];
  const float* gat_w     = (const float*)d_in[11];
  const float* gat_asrc  = (const float*)d_in[12];
  const float* gat_adst  = (const float*)d_in[13];
  const float* gat_bias  = (const float*)d_in[14];
  const float* atom_w    = (const float*)d_in[15];
  const float* atom_b    = (const float*)d_in[16];
  const float* res_w     = (const float*)d_in[17];
  const float* res_b     = (const float*)d_in[18];
  const float* mol_w     = (const float*)d_in[19];
  const float* mol_asrc  = (const float*)d_in[20];
  const float* mol_adst  = (const float*)d_in[21];
  const float* mol_bias  = (const float*)d_in[22];
  const float* gru_wih   = (const float*)d_in[23];
  const float* gru_whh   = (const float*)d_in[24];
  const float* gru_bih   = (const float*)d_in[25];
  const float* gru_bhh   = (const float*)d_in[26];
  const float* lin2_w    = (const float*)d_in[27];
  const float* lin2_b    = (const float*)d_in[28];
  float* out = (float*)d_out;          // [G] pred then [G*C] fp

  // ---- workspace bump allocator ----
  char* wp = (char*)d_ws;
  auto alloc = [&](size_t bytes) -> void* {
    void* r = (void*)wp;
    wp += (bytes + 255) & ~(size_t)255;
    return r;
  };
  float*    x1     = (float*)alloc(NC * 4);   // also reused as x3
  float*    res0   = (float*)alloc(NC * 4);
  float*    x2     = (float*)alloc(NC * 4);
  float*    ntmp   = (float*)alloc(NC * 4);
  float*    xw     = (float*)alloc(NC * 4);   // xw then xs
  float*    mnode  = (float*)alloc(NC * 4);
  _Float16* xj     = (_Float16*)alloc(EC * 2);
  float*    ebuf   = (float*)alloc((long)Ee * 4);
  float*    el     = (float*)alloc((long)Ee * 4);
  float*    dr     = (float*)alloc((long)Nn * 4);
  float*    s_src  = (float*)alloc((long)Nn * 4);
  float*    s_dst  = (float*)alloc((long)Nn * 4);
  float*    asrc   = (float*)alloc((long)Nn * 4);
  float*    nlog   = (float*)alloc((long)Nn * 4);
  unsigned* segmax = (unsigned*)alloc((long)Nn * 4);
  float*    segsum = (float*)alloc((long)Nn * 4);
  float*    outg   = (float*)alloc(GC * 4);
  float*    outw   = (float*)alloc(GC * 4);
  float*    hbuf   = (float*)alloc(GC * 4);
  float*    gi     = (float*)alloc((long)Gg * 3 * Cc * 4);
  float*    gh     = (float*)alloc((long)Gg * 3 * Cc * 4);
  float*    adst   = (float*)alloc((long)Gg * 4);
  // packed weights: size = COUT * KP halfs
  _Float16* wpl1   = (_Float16*)alloc((long)Cc * 64 * 2);
  _Float16* wpres  = (_Float16*)alloc((long)Cc * 128 * 2);
  _Float16* wpg1   = (_Float16*)alloc((long)Cc * 160 * 2);
  _Float16* wpg2   = (_Float16*)alloc((long)Cc * 128 * 2);
  _Float16* wpgat  = (_Float16*)alloc((long)Cc * 128 * 2);
  _Float16* wpatom = (_Float16*)alloc((long)Cc * 128 * 2);
  _Float16* wpmol  = (_Float16*)alloc((long)Cc * 128 * 2);
  _Float16* wpih   = (_Float16*)alloc((long)3 * Cc * 128 * 2);
  _Float16* wphh   = (_Float16*)alloc((long)3 * Cc * 128 * 2);

  auto nb = [](long n, int b) { return (unsigned)((n + b - 1) / b); };
  auto packw = [&](const float* W, _Float16* Wpk, int K, int KP, int COUT) {
    long total = (long)COUT * KP;
    k_packw<<<nb(total, 256), 256, 0, stream>>>(W, Wpk, K, KP, COUT);
  };
  auto gemm = [&](int K, int xmode, const void* X, const int* gidx, const float* ea,
                  const _Float16* Wpk, const float* b, float* Yf, _Float16* Yh,
                  const int* sidx, const float* sc, int M, int COUT, int act) {
    dim3 grid(M / 16, COUT / 128);
    if (K == 64)
      k_wmma_gemm<64, 0><<<grid, 128, 0, stream>>>(X, gidx, ea, Wpk, b, Yf, Yh, sidx, sc, M, COUT, act);
    else if (K == 144)
      k_wmma_gemm<144, 2><<<grid, 128, 0, stream>>>(X, gidx, ea, Wpk, b, Yf, Yh, sidx, sc, M, COUT, act);
    else if (xmode == 1)
      k_wmma_gemm<128, 1><<<grid, 128, 0, stream>>>(X, gidx, ea, Wpk, b, Yf, Yh, sidx, sc, M, COUT, act);
    else
      k_wmma_gemm<128, 0><<<grid, 128, 0, stream>>>(X, gidx, ea, Wpk, b, Yf, Yh, sidx, sc, M, COUT, act);
  };
  auto rowdot = [&](const void* X, int isHalf, const float* v, const float* b,
                    float* o, int M, int K) {
    k_rowdot<<<nb((long)M * 32, 256), 256, 0, stream>>>(X, isHalf, v, b, o, M, K);
  };
  auto segsoftmax = [&](float* logit, const int* seg, int nitems, int nsegs) {
    k_fill_u32<<<nb(nsegs, 256), 256, 0, stream>>>(segmax, NEG_INF_ORD, nsegs);
    k_fill_f32<<<nb(nsegs, 256), 256, 0, stream>>>(segsum, 0.f, nsegs);
    k_seg_max <<<nb(nitems, 256), 256, 0, stream>>>(logit, seg, segmax, nitems);
    k_seg_exp <<<nb(nitems, 256), 256, 0, stream>>>(logit, seg, segmax, segsum, nitems);
    k_seg_norm<<<nb(nitems, 256), 256, 0, stream>>>(logit, seg, segsum, nitems);
  };

  // ---- pack all weights into B-fragment layout ----
  packw(lin1_w, wpl1, INd, 64, Cc);
  packw(res_w, wpres, Cc, 128, Cc);
  packw(gate_l1_w, wpg1, Cc + EDd, 160, Cc);
  packw(gate_l2_w, wpg2, Cc, 128, Cc);
  packw(gat_w, wpgat, Cc, 128, Cc);
  packw(atom_w, wpatom, Cc, 128, Cc);
  packw(mol_w, wpmol, Cc, 128, Cc);
  packw(gru_wih, wpih, Cc, 128, 3 * Cc);
  packw(gru_whh, wphh, Cc, 128, 3 * Cc);

  // ---- x1 = lrelu(x0 @ lin1_w + b); res0 = lrelu(x1 @ res_w + b) ----
  gemm(64, 0, x0, nullptr, nullptr, wpl1, lin1_b, x1, nullptr, nullptr, nullptr, Nn, Cc, 1);
  gemm(128, 0, x1, nullptr, nullptr, wpres, res_b, res0, nullptr, nullptr, nullptr, Nn, Cc, 1);

  // ---- GATEConv ----
  gemm(144, 2, x1, src, edge_attr, wpg1, nullptr, nullptr, xj, nullptr, nullptr, Ee, Cc, 1);
  rowdot(xj, 1, gate_att_l, nullptr, el, Ee, Cc);
  rowdot(x1, 0, gate_att_r, nullptr, dr, Nn, Cc);
  k_logit<<<nb(Ee, 256), 256, 0, stream>>>(el, nullptr, dr, dst, ebuf, Ee);
  segsoftmax(ebuf, dst, Ee, Nn);
  k_fill_f32<<<nb(NC, 256), 256, 0, stream>>>(mnode, 0.f, (int)NC);
  gemm(128, 1, xj, nullptr, nullptr, wpg2, nullptr, mnode, nullptr, dst, ebuf, Ee, Cc, 0);
  gemm(128, 0, x1, nullptr, nullptr, wpatom, atom_b, ntmp, nullptr, nullptr, nullptr, Nn, Cc, 1);
  k_combine<<<nb(NC, 256), 256, 0, stream>>>(ntmp, mnode, gate_bias, nullptr, x2, NC, Cc);

  // ---- GATConv layer 2 ----
  gemm(128, 0, x2, nullptr, nullptr, wpgat, nullptr, xw, nullptr, nullptr, nullptr, Nn, Cc, 0);
  rowdot(xw, 0, gat_asrc, nullptr, s_src, Nn, Cc);
  rowdot(xw, 0, gat_adst, nullptr, s_dst, Nn, Cc);
  k_logit<<<nb(Ee, 256), 256, 0, stream>>>(s_src, src, s_dst, dst, ebuf, Ee);
  segsoftmax(ebuf, dst, Ee, Nn);
  k_fill_f32<<<nb(NC, 256), 256, 0, stream>>>(mnode, 0.f, (int)NC);
  k_scatter_rows<<<nb(EC, 256), 256, 0, stream>>>(xw, src, ebuf, dst, mnode, Ee, Cc);
  gemm(128, 0, x2, nullptr, nullptr, wpatom, atom_b, ntmp, nullptr, nullptr, nullptr, Nn, Cc, 1);
  k_combine<<<nb(NC, 256), 256, 0, stream>>>(ntmp, mnode, gat_bias, res0, x1, NC, Cc); // x3 -> x1

  // ---- molecule readout ----
  k_fill_f32<<<nb(GC, 256), 256, 0, stream>>>(outg, 0.f, (int)GC);
  k_scatter_rows<<<nb(NC, 256), 256, 0, stream>>>(x1, nullptr, nullptr, batch, outg, Nn, Cc);
  k_relu_ip<<<nb(GC, 256), 256, 0, stream>>>(outg, GC);
  gemm(128, 0, x1, nullptr, nullptr, wpmol, nullptr, xw, nullptr, nullptr, nullptr, Nn, Cc, 0); // xs
  rowdot(xw, 0, mol_asrc, nullptr, asrc, Nn, Cc);

  for (int pass = 0; pass < 2; ++pass) {
    gemm(128, 0, outg, nullptr, nullptr, wpmol, nullptr, outw, nullptr, nullptr, nullptr, Gg, Cc, 0);
    rowdot(outw, 0, mol_adst, nullptr, adst, Gg, Cc);
    k_logit<<<nb(Nn, 256), 256, 0, stream>>>(asrc, nullptr, adst, batch, nlog, Nn);
    segsoftmax(nlog, batch, Nn, Gg);
    k_fill_f32<<<nb(GC, 256), 256, 0, stream>>>(hbuf, 0.f, (int)GC);
    k_scatter_rows<<<nb(NC, 256), 256, 0, stream>>>(xw, nullptr, nlog, batch, hbuf, Nn, Cc);
    k_elu_bias_ip<<<nb(GC, 256), 256, 0, stream>>>(hbuf, mol_bias, GC, Cc);
    gemm(128, 0, hbuf, nullptr, nullptr, wpih, gru_bih, gi, nullptr, nullptr, nullptr, Gg, 3 * Cc, 0);
    gemm(128, 0, outg, nullptr, nullptr, wphh, gru_bhh, gh, nullptr, nullptr, nullptr, Gg, 3 * Cc, 0);
    k_gru<<<nb(GC, 256), 256, 0, stream>>>(gi, gh, outg, Gg, Cc);
  }

  // ---- final: pred = out @ lin2_w + b ; fp = out ----
  rowdot(outg, 0, lin2_w, lin2_b, out, Gg, Cc);
  k_copy<<<nb(GC, 256), 256, 0, stream>>>(outg, out + Gg, GC);
  (void)in_sizes; (void)n_in; (void)out_size; (void)ws_size;
}
